// LIF_68006512165374
// MI455X (gfx1250) — matrix operand order repaired and checked by reading, not compiled, not run on hardware
//
#include <hip/hip_runtime.h>
#include <cstdint>
#include <cstddef>

typedef __bf16 bf16x4  __attribute__((ext_vector_type(4)));
typedef __bf16 bf16x8  __attribute__((ext_vector_type(8)));
typedef __bf16 bf16x16 __attribute__((ext_vector_type(16)));
typedef float  f32x4   __attribute__((ext_vector_type(4)));
typedef float  f32x8   __attribute__((ext_vector_type(8)));

#define THR_V    0.6f
#define T_REFRAC 2.0f
#define DECAY    0.95122942450071400910f

#define N_B     128
#define T_STEPS 500
#define KDIM    1024   // n_in == n_rec == 1024
#define NREC    1024
#define MTOT    (N_B * T_STEPS)   // 64000 rows in batched input GEMM

// ---------- helpers ----------
static __device__ __forceinline__ bf16x16 cat8(bf16x8 a, bf16x8 b) {
  return __builtin_shufflevector(a, b, 0,1,2,3,4,5,6,7,8,9,10,11,12,13,14,15);
}
static __device__ __forceinline__ f32x8 wmma_bf16(bf16x16 a, bf16x16 b, f32x8 c) {
  // D = A(16x32 bf16) * B(32x16 bf16) + C(16x16 f32)
  return __builtin_amdgcn_wmma_f32_16x16x32_bf16(false, a, false, b, (short)0, c,
                                                 false, false);
}
static __device__ __forceinline__ __bf16 bits_to_bf(unsigned short h) {
  return __builtin_bit_cast(__bf16, h);
}
static __device__ __forceinline__ __bf16 f32_to_bf_rne(float f) {
  unsigned u = __builtin_bit_cast(unsigned, f);
  unsigned r = u + 0x7FFFu + ((u >> 16) & 1u);
  return bits_to_bf((unsigned short)(r >> 16));
}
// split f = hi + lo, hi = truncated-top-16 bf16, lo = RNE(remainder)
static __device__ __forceinline__ void split_bf16(float f, __bf16* hi, __bf16* lo) {
  unsigned u  = __builtin_bit_cast(unsigned, f) & 0xFFFF0000u;
  float   hf  = __builtin_bit_cast(float, u);
  *hi = bits_to_bf((unsigned short)(u >> 16));
  *lo = f32_to_bf_rne(f - hf);
}
static __device__ __forceinline__ float bf_to_f32(__bf16 h) {
  unsigned u = ((unsigned)__builtin_bit_cast(unsigned short, h)) << 16;
  return __builtin_bit_cast(float, u);
}
// CDNA5 async copy: 16B global -> LDS, tracked by ASYNCcnt (GVS addressing)
static __device__ __forceinline__ void async_cp16(unsigned lds_addr, unsigned voff,
                                                  const __bf16* sbase) {
  asm volatile("global_load_async_to_lds_b128 %0, %1, %2"
               :: "v"(lds_addr), "v"(voff),
                  "s"((unsigned long long)(uintptr_t)sbase)
               : "memory");
}

// ---------- x prep: f32 [M,K] -> bf16 hi/lo planes ----------
__global__ __launch_bounds__(256) void k_conv_x(const float* __restrict__ X,
                                                __bf16* __restrict__ Xh,
                                                __bf16* __restrict__ Xl) {
  size_t idx = ((size_t)blockIdx.x * 256 + threadIdx.x) * 4;  // grid = 64000 blocks
  f32x4 a = __builtin_nontemporal_load((const f32x4*)(X + idx));
  bf16x4 h, l;
#pragma unroll
  for (int j = 0; j < 4; ++j) {
    __bf16 hh, ll;
    split_bf16(a[j], &hh, &ll);
    h[j] = hh; l[j] = ll;
  }
  *(bf16x4*)(Xh + idx) = h;
  *(bf16x4*)(Xl + idx) = l;
}

// ---------- weight prep: f32 [K,N] -> bf16 hi/lo transposed [N,K] ----------
__global__ __launch_bounds__(256) void k_conv_w(const float* __restrict__ W,
                                                __bf16* __restrict__ Th,
                                                __bf16* __restrict__ Tl) {
  int idx = blockIdx.x * 256 + threadIdx.x;   // grid sized to exactly 1024*1024
  int k = idx >> 10, n = idx & 1023;
  __bf16 hi, lo;
  split_bf16(W[idx], &hi, &lo);
  Th[(size_t)n * KDIM + k] = hi;
  Tl[(size_t)n * KDIM + k] = lo;
}

// ---------- zero state ----------
__global__ __launch_bounds__(256) void k_init(__bf16* __restrict__ z0,
                                              float* __restrict__ v,
                                              float* __restrict__ r) {
  int idx = blockIdx.x * 256 + threadIdx.x;   // grid sized to 128*1024
  z0[idx] = bits_to_bf(0);
  v[idx] = 0.0f;
  r[idx] = 0.0f;
}

// ---------- phase 1: i_inx = x @ w_in  (bf16x3 split WMMA) ----------
// 256 thr (8 waves). Block tile 64M x 128N; wave = 16M x 64N (4 n-tiles).
// A (pre-split bf16 hi/lo) staged via double-buffered async LDS copies.
// grid.x = N tiles (fast) so the 8 blocks sharing one A tile run back-to-back.
__global__ __launch_bounds__(256) void k_gemm_in(const __bf16* __restrict__ Xh,
                                                 const __bf16* __restrict__ Xl,
                                                 const __bf16* __restrict__ Bh,
                                                 const __bf16* __restrict__ Bl,
                                                 float* __restrict__ C) {
  __shared__ alignas(16) __bf16 sA[2][2][64][32];   // [buf][hi/lo][row][k] = 16 KB
  const int tid  = threadIdx.x;
  const int wave = tid >> 5, lane = tid & 31;
  const int nblk = blockIdx.x * 128;                 // N tile (fast dim)
  const size_t mblk = (size_t)blockIdx.y * 64;       // M tile
  const int wm = (wave >> 1) * 16;
  const int wn = (wave & 1) * 64;
  const int half = lane >> 4, lm = lane & 15;
  const int ak0 = half ? 8 : 0;                      // A-frag K chunks (ISA 7.12.2)
  const int ak1 = half ? 24 : 16;
  const int bk  = half ? 8 : 0;                      // B-frag: K chunks bk, bk+16 (match A)

  // async staging: 256 threads cover 64 rows x 4 x 16B segments
  const int srow = tid >> 2;
  const int scol = (tid & 3) * 8;                    // element offset in row
  unsigned ldsH[2], ldsL[2];
  ldsH[0] = (unsigned)(uintptr_t)&sA[0][0][srow][scol];
  ldsH[1] = (unsigned)(uintptr_t)&sA[1][0][srow][scol];
  ldsL[0] = (unsigned)(uintptr_t)&sA[0][1][srow][scol];
  ldsL[1] = (unsigned)(uintptr_t)&sA[1][1][srow][scol];
  unsigned vof = (unsigned)(((mblk + srow) * (size_t)KDIM + scol) * 2u); // bytes

  f32x8 acc[4] = {{0.f},{0.f},{0.f},{0.f}};

  // prologue: chunk 0 -> buffer 0
  async_cp16(ldsH[0], vof, Xh);
  async_cp16(ldsL[0], vof, Xl);
  unsigned vofn = vof + 64;                          // next chunk (32 bf16 = 64 B)

  for (int k0 = 0; k0 < KDIM; k0 += 32) {
    const int buf = (k0 >> 5) & 1;
    if (k0 + 32 < KDIM) {                            // issue next chunk, keep 2 in flight
      async_cp16(ldsH[buf ^ 1], vofn, Xh);
      async_cp16(ldsL[buf ^ 1], vofn, Xl);
      vofn += 64;
      asm volatile("s_wait_asynccnt 0x2" ::: "memory");
    } else {
      asm volatile("s_wait_asynccnt 0x0" ::: "memory");
    }
    __syncthreads();                                 // chunk k visible to all waves

    bf16x16 Ah = cat8(*(const bf16x8*)&sA[buf][0][wm + lm][ak0],
                      *(const bf16x8*)&sA[buf][0][wm + lm][ak1]);
    bf16x16 Al = cat8(*(const bf16x8*)&sA[buf][1][wm + lm][ak0],
                      *(const bf16x8*)&sA[buf][1][wm + lm][ak1]);

    const __bf16* bhp = Bh + (size_t)(nblk + wn + lm) * KDIM + k0 + bk;
    const __bf16* blp = Bl + (size_t)(nblk + wn + lm) * KDIM + k0 + bk;
#pragma unroll
    for (int nt = 0; nt < 4; ++nt) {
      const __bf16* bh = bhp + (size_t)nt * 16 * KDIM;
      const __bf16* bl = blp + (size_t)nt * 16 * KDIM;
      bf16x16 Bhv = cat8(*(const bf16x8*)bh, *(const bf16x8*)(bh + 16));
      bf16x16 Blv = cat8(*(const bf16x8*)bl, *(const bf16x8*)(bl + 16));
      acc[nt] = wmma_bf16(Ah, Bhv, acc[nt]);   // hi*hi
      acc[nt] = wmma_bf16(Al, Bhv, acc[nt]);   // lo*hi
      acc[nt] = wmma_bf16(Ah, Blv, acc[nt]);   // hi*lo
    }
    __syncthreads();                                 // release buf before overwrite
  }

  const size_t crow0 = mblk + wm + half * 8;   // C layout: VGPR e -> M = e + 8*half
#pragma unroll
  for (int nt = 0; nt < 4; ++nt) {
    const int ncol = nblk + wn + nt * 16 + lm;
#pragma unroll
    for (int e = 0; e < 8; ++e)
      __builtin_nontemporal_store(acc[nt][e], C + (crow0 + e) * (size_t)NREC + ncol);
  }
}

// ---------- phase 2 (per step): z@w_rec + LIF update ----------
// 256 thr (8 waves) as 2(M)x4(N); block tile 32M x 64N; grid (4,16).
// z is exact in bf16 -> only 2 wmmas per k-chunk (z*Whi + z*Wlo).
__global__ __launch_bounds__(256) void k_step(const __bf16* __restrict__ Zin,
                                              __bf16*       __restrict__ Zout,
                                              const __bf16* __restrict__ Wh,
                                              const __bf16* __restrict__ Wl,
                                              const float*  __restrict__ Iin,
                                              float* __restrict__ Vbuf,
                                              float* __restrict__ Rbuf,
                                              float* __restrict__ Zs,
                                              float* __restrict__ Vs,
                                              int t) {
  const int tid  = threadIdx.x;
  const int wave = tid >> 5, lane = tid & 31;
  const int half = lane >> 4, lm = lane & 15;
  const int mbase = blockIdx.x * 32 + ((wave >> 2) & 1) * 16;
  const int nbase = blockIdx.y * 64 + (wave & 3) * 16;
  const int ak0 = half ? 8 : 0, ak1 = half ? 24 : 16;
  const int bk  = half ? 8 : 0;

  const __bf16* zrow = Zin + (size_t)(mbase + lm) * NREC;
  const __bf16* whp  = Wh  + (size_t)(nbase + lm) * NREC + bk;
  const __bf16* wlp  = Wl  + (size_t)(nbase + lm) * NREC + bk;

  f32x8 acc = {0.f};
  for (int k0 = 0; k0 < NREC; k0 += 32) {
    bf16x16 Az  = cat8(*(const bf16x8*)(zrow + k0 + ak0),
                       *(const bf16x8*)(zrow + k0 + ak1));
    bf16x16 Bhv = cat8(*(const bf16x8*)(whp + k0), *(const bf16x8*)(whp + k0 + 16));
    bf16x16 Blv = cat8(*(const bf16x8*)(wlp + k0), *(const bf16x8*)(wlp + k0 + 16));
    acc = wmma_bf16(Az, Bhv, acc);
    acc = wmma_bf16(Az, Blv, acc);
  }

  const int n = nbase + lm;
#pragma unroll
  for (int e = 0; e < 8; ++e) {
    const int b = mbase + half * 8 + e;
    const size_t sidx = (size_t)b * NREC + n;                       // state [B,N]
    const size_t oidx = ((size_t)b * T_STEPS + t) * NREC + n;       // out   [B,T,N]
    float ii = acc[e] + __builtin_nontemporal_load(Iin + oidx);
    float v  = Vbuf[sidx];
    float rf = Rbuf[sidx];
    float zp = bf_to_f32(Zin[sidx]);
    float nv = DECAY * v + ii - zp * THR_V;
    float spike = (nv > THR_V) ? 1.0f : 0.0f;   // sign of (nv-THR)/THR
    float nz = (rf >= 1.0f) ? 0.0f : spike;
    float nr = rf + T_REFRAC * zp - 1.0f;
    nr = fminf(fmaxf(nr, 0.0f), T_REFRAC);
    Vbuf[sidx] = nv;
    Rbuf[sidx] = nr;
    Zout[sidx] = bits_to_bf(nz > 0.5f ? (unsigned short)0x3F80 : (unsigned short)0);
    __builtin_nontemporal_store(nz, Zs + oidx);
    __builtin_nontemporal_store(nv, Vs + oidx);
  }
}

// ---------- host ----------
extern "C" void kernel_launch(void* const* d_in, const int* in_sizes, int n_in,
                              void* d_out, int out_size, void* d_ws, size_t ws_size,
                              hipStream_t stream) {
  (void)in_sizes; (void)n_in; (void)out_size; (void)ws_size;
  const float* x     = (const float*)d_in[0];
  const float* w_in  = (const float*)d_in[1];
  const float* w_rec = (const float*)d_in[2];

  char* ws = (char*)d_ws;
  size_t off = 0;
  auto take = [&](size_t bytes) -> char* {
    char* p = ws + off;
    off = (off + bytes + 255) & ~(size_t)255;
    return p;
  };
  float*  iinx   = (float*) take((size_t)MTOT * NREC * sizeof(float));
  __bf16* winTh  = (__bf16*)take((size_t)KDIM * NREC * sizeof(__bf16));
  __bf16* winTl  = (__bf16*)take((size_t)KDIM * NREC * sizeof(__bf16));
  __bf16* wrecTh = (__bf16*)take((size_t)NREC * NREC * sizeof(__bf16));
  __bf16* wrecTl = (__bf16*)take((size_t)NREC * NREC * sizeof(__bf16));
  __bf16* zb0    = (__bf16*)take((size_t)N_B * NREC * sizeof(__bf16));
  __bf16* zb1    = (__bf16*)take((size_t)N_B * NREC * sizeof(__bf16));
  float*  vbuf   = (float*) take((size_t)N_B * NREC * sizeof(float));
  float*  rbuf   = (float*) take((size_t)N_B * NREC * sizeof(float));

  float* zs = (float*)d_out;
  float* vs = zs + (size_t)N_B * T_STEPS * NREC;

  // Stash pre-split x (bf16 hi/lo, 2 x 128 MB) in d_out's space: phase 1 consumes
  // it before phase 2 overwrites every element of d_out (stream-ordered).
  __bf16* xh = (__bf16*)d_out;
  __bf16* xl = xh + (size_t)MTOT * KDIM;

  // x split (64M elems, 4 per thread)
  k_conv_x<<<MTOT * KDIM / 1024, 256, 0, stream>>>(x, xh, xl);
  // weight split + transpose (1M elems each)
  k_conv_w<<<4096, 256, 0, stream>>>(w_in,  winTh,  winTl);
  k_conv_w<<<4096, 256, 0, stream>>>(w_rec, wrecTh, wrecTl);
  // zero state (128*1024 elems)
  k_init<<<512, 256, 0, stream>>>(zb0, vbuf, rbuf);
  // batched input GEMM: [64000,1024] x [1024,1024]; N tiles on fast grid dim
  k_gemm_in<<<dim3(NREC / 128, MTOT / 64), 256, 0, stream>>>(
      xh, xl, winTh, winTl, iinx);
  // sequential recurrent scan, ping-pong z buffers
  __bf16* zb[2] = {zb0, zb1};
  for (int t = 0; t < T_STEPS; ++t) {
    k_step<<<dim3(N_B / 32, NREC / 64), 256, 0, stream>>>(
        zb[t & 1], zb[(t + 1) & 1], wrecTh, wrecTl, iinx, vbuf, rbuf, zs, vs, t);
  }
}
